// CATLayer_75660143886733
// MI455X (gfx1250) — matrix operand order repaired and checked by reading, not compiled, not run on hardware
//
#include <hip/hip_runtime.h>

// ---------------- types ----------------
typedef _Float16 half_t;
typedef __attribute__((ext_vector_type(16))) half_t v16h;
typedef __attribute__((ext_vector_type(8)))  float  v8f;

constexpr int Dm   = 512;    // embed
constexpr int Fm   = 2048;   // 4*D
constexpr int NTOK = 4096;   // B*S
constexpr int EXP  = 8;

// padded LDS strides (halves) to avoid 16-way bank conflicts on ds_load_b128:
// stride*2 bytes /4 % 64 must be a small non-zero multiple of 4.
constexpr int HLD = Fm + 8;  // 2056: hidden tile stride  (4112B -> bank base 4*m)
constexpr int XLD = Dm + 8;  // 520 : x tile / attn LDS stride (1040B -> bank base 4*m)

// ---------------- fragment loaders (CDNA5 WMMA 16x16x32 f16 layouts) ----------------
// A matrix 16x32 (MxK), row-major source with leading dim ld:
//   lane m = lane&15 ; lanes<16 hold K 0..7 & 16..23 ; lanes>=16 hold K 8..15 & 24..31
// => two contiguous 16B loads per lane.
__device__ __forceinline__ v16h load_a_frag(const half_t* p, int ld, int lane) {
  int m  = lane & 15;
  int kh = (lane >> 4) << 3;              // 0 or 8
  const half_t* r = p + m * ld + kh;
  union { uint4 u[2]; v16h v; } u;
  u.u[0] = *(const uint4*)(r);
  u.u[1] = *(const uint4*)(r + 16);
  return u.v;
}

// B matrix 32x16 (KxN), pre-swizzled fragment-major: lane gets 16 contiguous halves.
__device__ __forceinline__ v16h load_b_frag(const half_t* p, int lane) {
  const uint4* q = (const uint4*)(p + (lane << 4));
  union { uint4 u[2]; v16h v; } u;
  u.u[0] = q[0];
  u.u[1] = q[1];
  return u.v;
}

__device__ __forceinline__ v8f wmma_f16(v16h a, v16h b, v8f c) {
  // (neg_a, A, neg_b, B, c_mod, C, reuse_a, reuse_b)
  return __builtin_amdgcn_wmma_f32_16x16x32_f16(false, a, false, b, (short)0, c,
                                                false, false);
}

// ---------------- pre-pass: fp32 -> f16 elementwise ----------------
__global__ __launch_bounds__(256) void k_cvt(const float* __restrict__ src,
                                             half_t* __restrict__ dst, long n) {
  long stride = (long)gridDim.x * blockDim.x;
  for (long i = (long)blockIdx.x * blockDim.x + threadIdx.x; i < n; i += stride)
    dst[i] = (half_t)src[i];
}

// ---------------- pre-pass: swizzle fp32 matrix -> f16 B-fragment-major ----------------
// Logical B is [K,N] (K mult of 32, N mult of 16). transposed=1 => B(k,n)=src[n*K+k].
// Dst tile t = nt*(K/32)+kt, 512 halves/tile; lane l stores
// B(kt*32 + (l<16?0:8) + (p&7) + (p>=8?16:0), nt*16 + (l&15)) at position l*16+p.
__global__ __launch_bounds__(256) void k_swizzleB(const float* __restrict__ src,
                                                  half_t* __restrict__ dst,
                                                  int K, int N, int nmat, int transposed) {
  long perMat = (long)K * N;
  long total  = perMat * nmat;
  int ktiles  = K >> 5;
  long stride = (long)gridDim.x * blockDim.x;
  for (long i = (long)blockIdx.x * blockDim.x + threadIdx.x; i < total; i += stride) {
    long mi = i / perMat;
    long il = i - mi * perMat;
    int t = (int)(il >> 9);
    int w = (int)(il & 511);
    int l = w >> 4, p = w & 15;
    int nt = t / ktiles, kt = t - nt * ktiles;
    int k = (kt << 5) + ((l >> 4) << 3) + (p & 7) + ((p >> 3) << 4);
    int n = (nt << 4) + (l & 15);
    long sidx = mi * perMat + (transposed ? ((long)n * K + k) : ((long)k * N + n));
    dst[i] = (half_t)src[sidx];
  }
}

// ---------------- fused per-expert FFN: a = relu(x@W1+b1)@W2+b2 ----------------
// grid = 1024 (128 token-tiles of 32 rows x 8 experts), block = 256 (8 waves).
// dynamic LDS: hidden [32][2056] f16 (128.5KB) + x tile [32][520] f16 (32.5KB).
__global__ __launch_bounds__(256) void k_ffn(
    const half_t* __restrict__ xb, const half_t* __restrict__ w1s,
    const float* __restrict__ b1, const half_t* __restrict__ w2s,
    const float* __restrict__ b2, half_t* __restrict__ abuf) {
  extern __shared__ char smem[];
  half_t* hl = (half_t*)smem;                       // [32][HLD]
  half_t* xl = (half_t*)(smem + 32 * HLD * 2);      // [32][XLD]
  int tile = blockIdx.x & 127;
  int e    = blockIdx.x >> 7;
  int wid  = threadIdx.x >> 5;
  int lane = threadIdx.x & 31;
  int mt   = wid & 1;                        // M sub-tile (0/1)
  int cg   = wid >> 1;                       // column group 0..3
  const half_t* w1e = w1s + (long)e * Dm * Fm;
  const half_t* w2e = w2s + (long)e * Fm * Dm;
  int colL  = lane & 15;
  int mbase = (lane >> 4) << 3;

  // ---- stage x tile [32,512] into LDS (coalesced b128), padded stride ----
  {
    const half_t* xsrc = xb + (long)tile * 32 * Dm;
    int tid = threadIdx.x;
#pragma unroll
    for (int it = 0; it < 8; ++it) {
      int idx = it * 2048 + tid * 8;         // 8 halves per thread
      int row = idx >> 9;
      int col = idx & 511;
      *(uint4*)(xl + row * XLD + col) = *(const uint4*)(xsrc + idx);
    }
  }
  __syncthreads();

  // ---- GEMM1: hidden[mt*16..+16, cg*512..+512] = relu(x@W1 + b1) ----
  const half_t* xrow = xl + mt * 16 * XLD;
#pragma unroll 1
  for (int g = 0; g < 8; ++g) {
    v8f acc[4] = {};
    int nt0 = cg * 32 + g * 4;
#pragma unroll 1
    for (int kt = 0; kt < 16; ++kt) {
      v16h a = load_a_frag(xrow + kt * 32, XLD, lane);
      __builtin_prefetch((const void*)(w1e + (((long)(nt0 + 4) * 16 + kt) << 9) +
                                       (lane << 4)), 0, 1);
#pragma unroll
      for (int j = 0; j < 4; ++j) {
        v16h b = load_b_frag(w1e + (((long)(nt0 + j) * 16 + kt) << 9), lane);
        acc[j] = wmma_f16(a, b, acc[j]);
      }
    }
#pragma unroll
    for (int j = 0; j < 4; ++j) {
      int col = (nt0 + j) * 16 + colL;
      float bb = b1[e * Fm + col];
#pragma unroll
      for (int r = 0; r < 8; ++r) {
        float f = acc[j][r] + bb;
        hl[(mt * 16 + mbase + r) * HLD + col] = (half_t)(f > 0.f ? f : 0.f);
      }
    }
  }
  __syncthreads();

  // ---- GEMM2: a[tile rows, cg*128..+128] = hidden@W2 + b2 ----
  v8f acc2[8] = {};
  int nt0 = cg * 8;
  const half_t* hrow = hl + mt * 16 * HLD;
#pragma unroll 1
  for (int kt = 0; kt < 64; ++kt) {
    v16h a = load_a_frag(hrow + kt * 32, HLD, lane);
    __builtin_prefetch((const void*)(w2e + (((long)nt0 * 64 + kt + 4) << 9) +
                                     (lane << 4)), 0, 1);
#pragma unroll
    for (int j = 0; j < 8; ++j) {
      v16h b = load_b_frag(w2e + (((long)(nt0 + j) * 64 + kt) << 9), lane);
      acc2[j] = wmma_f16(a, b, acc2[j]);
    }
  }
#pragma unroll
  for (int j = 0; j < 8; ++j) {
    int col = (nt0 + j) * 16 + colL;
    float bb = b2[e * Dm + col];
    long rowbase = (long)e * NTOK + tile * 32 + mt * 16 + mbase;
#pragma unroll
    for (int r = 0; r < 8; ++r)
      abuf[(rowbase + r) * Dm + col] = (half_t)(acc2[j][r] + bb);
  }
}

// ---------------- attention over expert axis + out projection ----------------
// grid = 256 (16 tokens each), block = 256 (8 waves; wave e owns expert e's K/V).
// dynamic LDS (padded stride 520): q 16.25KB | k 130KB | v 130KB | ctx 16.25KB
//   = 299520B total (<= 320KB/WG on CDNA5).
__global__ __launch_bounds__(256) void k_attn(
    const half_t* __restrict__ abuf,
    const half_t* __restrict__ wqs, const float* __restrict__ bq,
    const half_t* __restrict__ wks, const float* __restrict__ bk,
    const half_t* __restrict__ wvs, const float* __restrict__ bv,
    const half_t* __restrict__ wos, const float* __restrict__ bo,
    const int* __restrict__ eidp, float* __restrict__ outp) {
  extern __shared__ char smem[];
  constexpr int QSZ = 16 * XLD * 2;        // 16640 B
  constexpr int KSZ = 8 * 16 * XLD * 2;    // 133120 B
  half_t* qls   = (half_t*)smem;
  half_t* kls   = (half_t*)(smem + QSZ);
  half_t* vls   = (half_t*)(smem + QSZ + KSZ);
  half_t* ctxls = (half_t*)(smem + QSZ + 2 * KSZ);
  int tile = blockIdx.x;
  int wid  = threadIdx.x >> 5;
  int lane = threadIdx.x & 31;
  int eid  = *eidp;
  int colL  = lane & 15;
  int mbase = (lane >> 4) << 3;

  // ---- K,V projections for expert e = wid (each wave: full [16,512]) ----
  const half_t* ae = abuf + ((long)wid * NTOK + tile * 16) * Dm;
#pragma unroll 1
  for (int nt = 0; nt < 32; ++nt) {
    v8f ak = {}, av = {};
#pragma unroll 1
    for (int kt = 0; kt < 16; ++kt) {
      v16h a  = load_a_frag(ae + kt * 32, Dm, lane);
      v16h bK = load_b_frag(wks + (((long)(nt << 4) + kt) << 9), lane);
      v16h bV = load_b_frag(wvs + (((long)(nt << 4) + kt) << 9), lane);
      ak = wmma_f16(a, bK, ak);
      av = wmma_f16(a, bV, av);
    }
    int col = (nt << 4) + colL;
    float bkv = bk[col], bvv = bv[col];
#pragma unroll
    for (int r = 0; r < 8; ++r) {
      kls[(wid * 16 + mbase + r) * XLD + col] = (half_t)(ak[r] + bkv);
      vls[(wid * 16 + mbase + r) * XLD + col] = (half_t)(av[r] + bvv);
    }
  }
  // ---- Q projection only for expert eid (4 n-tiles per wave) ----
  const half_t* aq = abuf + ((long)eid * NTOK + tile * 16) * Dm;
#pragma unroll 1
  for (int j = 0; j < 4; ++j) {
    int nt = wid * 4 + j;
    v8f aqc = {};
#pragma unroll 1
    for (int kt = 0; kt < 16; ++kt) {
      v16h a = load_a_frag(aq + kt * 32, Dm, lane);
      v16h b = load_b_frag(wqs + (((long)(nt << 4) + kt) << 9), lane);
      aqc = wmma_f16(a, b, aqc);
    }
    int col = (nt << 4) + colL;
    float bb = bq[col];
#pragma unroll
    for (int r = 0; r < 8; ++r)
      qls[(mbase + r) * XLD + col] = (half_t)(aqc[r] + bb);
  }
  __syncthreads();

  // ---- softmax attention over E=8 experts, per (token, head) ----
  if (threadIdx.x < 128) {
    int tok = threadIdx.x >> 3, h = threadIdx.x & 7;
    const half_t* qr = qls + tok * XLD + h * 64;
    float qv[64];
#pragma unroll
    for (int d0 = 0; d0 < 64; ++d0) qv[d0] = (float)qr[d0];
    float s[8], mx = -3.0e38f;
#pragma unroll 1
    for (int f = 0; f < 8; ++f) {
      const half_t* kr = kls + (f * 16 + tok) * XLD + h * 64;
      float acc = 0.f;
#pragma unroll
      for (int d0 = 0; d0 < 64; ++d0) acc += qv[d0] * (float)kr[d0];
      // additive float tril mask (torch semantics reproduced faithfully)
      acc = acc * 0.125f + ((f <= eid) ? 1.0f : 0.0f);
      s[f] = acc;
      mx = fmaxf(mx, acc);
    }
    float sum = 0.f;
#pragma unroll
    for (int f = 0; f < 8; ++f) { s[f] = __expf(s[f] - mx); sum += s[f]; }
    float inv = 1.0f / sum;
#pragma unroll
    for (int f = 0; f < 8; ++f) s[f] *= inv;
#pragma unroll 1
    for (int d0 = 0; d0 < 64; ++d0) {
      float c = 0.f;
#pragma unroll
      for (int f = 0; f < 8; ++f)
        c += s[f] * (float)vls[(f * 16 + tok) * XLD + h * 64 + d0];
      ctxls[tok * XLD + h * 64 + d0] = (half_t)c;
    }
  }
  __syncthreads();

  // ---- out = ctx @ Wo.T + bo  (fp32 result) ----
#pragma unroll 1
  for (int j = 0; j < 4; ++j) {
    int nt = wid * 4 + j;
    v8f acc = {};
#pragma unroll 1
    for (int kt = 0; kt < 16; ++kt) {
      v16h a = load_a_frag(ctxls + kt * 32, XLD, lane);
      v16h b = load_b_frag(wos + (((long)(nt << 4) + kt) << 9), lane);
      acc = wmma_f16(a, b, acc);
    }
    int col = (nt << 4) + colL;
    float bb = bo[col];
#pragma unroll
    for (int r = 0; r < 8; ++r)
      outp[((long)tile * 16 + mbase + r) * Dm + col] = acc[r] + bb;
  }
}

// ---------------- host launcher ----------------
extern "C" void kernel_launch(void* const* d_in, const int* in_sizes, int n_in,
                              void* d_out, int out_size, void* d_ws, size_t ws_size,
                              hipStream_t stream) {
  const float* x  = (const float*)d_in[0];
  const float* W1 = (const float*)d_in[1];
  const float* b1 = (const float*)d_in[2];
  const float* W2 = (const float*)d_in[3];
  const float* b2 = (const float*)d_in[4];
  const float* Wq = (const float*)d_in[5];
  const float* bq = (const float*)d_in[6];
  const float* Wk = (const float*)d_in[7];
  const float* bk = (const float*)d_in[8];
  const float* Wv = (const float*)d_in[9];
  const float* bv = (const float*)d_in[10];
  const float* Wo = (const float*)d_in[11];
  const float* bo = (const float*)d_in[12];
  const int*  eid = (const int*)d_in[13];
  float* out = (float*)d_out;

  // scratch layout (bytes)
  char* ws = (char*)d_ws;
  const long OFF_XB = 0;                      // x f16:            4 MiB
  const long OFF_W1 = OFF_XB + 4194304;       // W1 swizzled f16: 16 MiB
  const long OFF_W2 = OFF_W1 + 16777216;      // W2 swizzled f16: 16 MiB
  const long OFF_WQ = OFF_W2 + 16777216;      // 512 KiB each
  const long OFF_WK = OFF_WQ + 524288;
  const long OFF_WV = OFF_WK + 524288;
  const long OFF_WO = OFF_WV + 524288;
  const long OFF_A  = OFF_WO + 524288;        // a f16 [E][N][D]: 32 MiB
  half_t* xb  = (half_t*)(ws + OFF_XB);
  half_t* w1s = (half_t*)(ws + OFF_W1);
  half_t* w2s = (half_t*)(ws + OFF_W2);
  half_t* wqs = (half_t*)(ws + OFF_WQ);
  half_t* wks = (half_t*)(ws + OFF_WK);
  half_t* wvs = (half_t*)(ws + OFF_WV);
  half_t* wos = (half_t*)(ws + OFF_WO);
  half_t* ab  = (half_t*)(ws + OFF_A);

  // pre-pass: convert + swizzle (weights fit in 192MB L2; re-streamed from L2 later)
  k_cvt<<<1024, 256, 0, stream>>>(x, xb, (long)NTOK * Dm);
  k_swizzleB<<<4096, 256, 0, stream>>>(W1, w1s, Dm, Fm, EXP, 0);
  k_swizzleB<<<4096, 256, 0, stream>>>(W2, w2s, Fm, Dm, EXP, 0);
  k_swizzleB<<<512,  256, 0, stream>>>(Wq, wqs, Dm, Dm, 1, 1);  // a @ Wq.T
  k_swizzleB<<<512,  256, 0, stream>>>(Wk, wks, Dm, Dm, 1, 1);
  k_swizzleB<<<512,  256, 0, stream>>>(Wv, wvs, Dm, Dm, 1, 1);
  k_swizzleB<<<512,  256, 0, stream>>>(Wo, wos, Dm, Dm, 1, 1);

  // fused per-expert FFN (dominant 137 GFLOP): 128 token-tiles x 8 experts
  k_ffn<<<1024, 256, 32 * HLD * 2 + 32 * XLD * 2, stream>>>(xb, w1s, b1, w2s, b2, ab);

  // attention over expert axis + output projection (292.5KB LDS per WG)
  k_attn<<<256, 256, (16 * XLD * 2) * 2 + (8 * 16 * XLD * 2) * 2, stream>>>(
      ab, wqs, bq, wks, bk, wvs, bv, wos, bo, eid, out);
}
